// LSTMClassifier_74062416052850
// MI455X (gfx1250) — compile-verified, matrix operand order checked by reading, not run on hardware
//
#include <hip/hip_runtime.h>

typedef _Float16 v16h __attribute__((ext_vector_type(16)));
typedef _Float16 v8h  __attribute__((ext_vector_type(8)));
typedef float    v8f  __attribute__((ext_vector_type(8)));

#define B_TOT 512
#define T_LEN 2048
#define IN    32
#define HID   32
#define GATES 128   // 4*H
#define NOUT  8

// Build WMMA B-tile (32xK -> 16 cols) for gate columns [g0, g0+16) from a
// (128,32) row-major fp32 weight, converting to f16.
// Lane layout mirrors the A layout: lane<16 -> col n=lane, K {0..7,16..23};
// lane>=16 -> col n=lane-16, K {8..15,24..31}.
__device__ inline v16h load_Btile(const float* __restrict__ W, int g0, int lane) {
    int n  = lane & 15;
    int c0 = (lane < 16) ? 0 : 8;
    const float* row = W + (size_t)(g0 + n) * IN;
    v16h b;
#pragma unroll
    for (int i = 0; i < 8; ++i) {
        b[i]     = (_Float16)row[c0 + i];
        b[i + 8] = (_Float16)row[c0 + 16 + i];
    }
    return b;
}

// Gather a 16x32 f16 LDS tile into WMMA A layout (two ds_load_b128 per lane).
__device__ inline v16h gather_A(const _Float16* hs, int lane) {
    int r  = lane & 15;
    int c0 = (lane < 16) ? 0 : 8;
    const v8h* plo = (const v8h*)(hs + r * 32 + c0);
    const v8h* phi = (const v8h*)(hs + r * 32 + c0 + 16);
    v8h lo = *plo;
    v8h hi = *phi;
    return __builtin_shufflevector(lo, hi, 0,1,2,3,4,5,6,7,8,9,10,11,12,13,14,15);
}

// CDNA5 has a native v_tanh_f32 TRANS op — use it; branch-free, co-executes
// with VALU. Sigmoid via sigma(x) = 0.5*tanh(x/2) + 0.5.
__device__ inline float fast_tanh(float v) {
#if __has_builtin(__builtin_amdgcn_tanhf)
    return __builtin_amdgcn_tanhf(v);
#else
    return tanhf(v);
#endif
}
__device__ inline float sigm(float v) { return __builtin_fmaf(0.5f, fast_tanh(0.5f * v), 0.5f); }

// One LSTM cell elementwise update for (row b, feature j); returns new c,
// writes new h (f16) into the A-layout staging tile.
__device__ inline float cellstep(const float* __restrict__ gs, int b, int j,
                                 float c, _Float16* __restrict__ hstage) {
    float ig = gs[b * GATES +       j];
    float fg = gs[b * GATES +  32 + j];
    float gg = gs[b * GATES +  64 + j];
    float og = gs[b * GATES +  96 + j];
    float i = sigm(ig);
    float f = sigm(fg);
    float g = fast_tanh(gg);
    float o = sigm(og);
    c = __builtin_fmaf(f, c, i * g);
    hstage[b * 32 + j] = (_Float16)(o * fast_tanh(c));
    return c;
}

__global__ __launch_bounds__(256)
void lstm2_fused_wmma(const float* __restrict__ x,
                      const float* __restrict__ Wih0, const float* __restrict__ Whh0,
                      const float* __restrict__ bih0, const float* __restrict__ bhh0,
                      const float* __restrict__ Wih1, const float* __restrict__ Whh1,
                      const float* __restrict__ bih1, const float* __restrict__ bhh1,
                      const float* __restrict__ Wfc,  const float* __restrict__ bfc,
                      float* __restrict__ out) {
    __shared__ __align__(16) _Float16 xs [16 * 32];   // x[t] tile (f16, row-major)
    __shared__ __align__(16) _Float16 h1s[16 * 32];   // h1 state (f16)
    __shared__ __align__(16) _Float16 h2s[16 * 32];   // h2 state (f16)
    __shared__ float gs[16 * GATES];                  // gate scratch (both layers)

    const int tid  = threadIdx.x;
    const int lane = tid & 31;
    const int wave = tid >> 5;
    const int b0   = blockIdx.x * 16;                 // batch tile origin

    // ---- per-wave constant weights in WMMA B layout (kept in VGPRs) ----
    const int g0 = wave * 16;                         // gate-column tile
    v16h B_ih0 = load_Btile(Wih0, g0, lane);
    v16h B_hh0 = load_Btile(Whh0, g0, lane);
    v16h B_ih1 = load_Btile(Wih1, g0, lane);
    v16h B_hh1 = load_Btile(Whh1, g0, lane);
    const int ncol = g0 + (lane & 15);
    const float bias1 = bih0[ncol] + bhh0[ncol];
    const float bias2 = bih1[ncol] + bhh1[ncol];

    // ---- init hidden-state staging tiles ----
    for (int e = tid; e < 16 * 32; e += 256) { h1s[e] = (_Float16)0.0f; h2s[e] = (_Float16)0.0f; }

    // cell states live in registers: this thread owns rows bA and bA+8, col jA
    const int bA = tid >> 5;          // 0..7
    const int jA = tid & 31;          // 0..31
    float c1a = 0.f, c1b = 0.f, c2a = 0.f, c2b = 0.f;

    // running per-thread stream pointers into x (bumped by IN floats per step)
    const float* px0 = x + ((size_t)(b0 + bA)     * T_LEN) * IN + jA;
    const float* px1 = x + ((size_t)(b0 + bA + 8) * T_LEN) * IN + jA;

    // preload x[t=0] tile into LDS
    xs[bA * 32 + jA]       = (_Float16)px0[0];
    xs[(bA + 8) * 32 + jA] = (_Float16)px1[0];
    __syncthreads();

    for (int t = 0; t < T_LEN; ++t) {
        // stream next timestep of x into registers; prefetch further ahead
        float nx0 = 0.f, nx1 = 0.f;
        if (t + 1 < T_LEN) { nx0 = px0[IN]; nx1 = px1[IN]; }
        if (t + 8 < T_LEN) {
            __builtin_prefetch(px0 + 8 * IN, 0, 3);
            __builtin_prefetch(px1 + 8 * IN, 0, 3);
        }
        px0 += IN; px1 += IN;

        // ---- layer 1 GEMM: gates = bias + x[t]@Wih0^T + h1@Whh0^T ----
        {
            v16h a_x = gather_A(xs,  lane);
            v16h a_h = gather_A(h1s, lane);
            v8f  c   = {bias1, bias1, bias1, bias1, bias1, bias1, bias1, bias1};
            c = __builtin_amdgcn_wmma_f32_16x16x32_f16(false, a_x, false, B_ih0, (short)0, c, false, false);
            c = __builtin_amdgcn_wmma_f32_16x16x32_f16(false, a_h, false, B_hh0, (short)0, c, false, false);
            const int n = lane & 15, mb = (lane < 16) ? 0 : 8;
#pragma unroll
            for (int r = 0; r < 8; ++r) gs[(mb + r) * GATES + g0 + n] = c[r];
        }
        __syncthreads();

        // ---- layer 1 elementwise (c in regs, h -> h1s as f16) ----
        c1a = cellstep(gs, bA,     jA, c1a, h1s);
        c1b = cellstep(gs, bA + 8, jA, c1b, h1s);
        __syncthreads();

        // ---- layer 2 GEMM: gates = bias + h1[t]@Wih1^T + h2@Whh1^T ----
        {
            v16h a_x = gather_A(h1s, lane);
            v16h a_h = gather_A(h2s, lane);
            v8f  c   = {bias2, bias2, bias2, bias2, bias2, bias2, bias2, bias2};
            c = __builtin_amdgcn_wmma_f32_16x16x32_f16(false, a_x, false, B_ih1, (short)0, c, false, false);
            c = __builtin_amdgcn_wmma_f32_16x16x32_f16(false, a_h, false, B_hh1, (short)0, c, false, false);
            const int n = lane & 15, mb = (lane < 16) ? 0 : 8;
#pragma unroll
            for (int r = 0; r < 8; ++r) gs[(mb + r) * GATES + g0 + n] = c[r];
        }
        __syncthreads();

        // ---- layer 2 elementwise + commit x[t+1] to LDS ----
        c2a = cellstep(gs, bA,     jA, c2a, h2s);
        c2b = cellstep(gs, bA + 8, jA, c2b, h2s);
        if (t + 1 < T_LEN) {
            xs[bA * 32 + jA]       = (_Float16)nx0;
            xs[(bA + 8) * 32 + jA] = (_Float16)nx1;
        }
        __syncthreads();
    }

    // ---- final FC: out[b,:] = h2_last[b,:] @ Wfc^T + bfc  (16x32 @ 32x8) ----
    if (tid < 16 * NOUT) {
        const int b = tid >> 3;     // 0..15
        const int o = tid & 7;      // 0..7
        float acc = bfc[o];
#pragma unroll
        for (int k = 0; k < HID; ++k)
            acc += (float)h2s[b * 32 + k] * Wfc[o * HID + k];
        out[(size_t)(b0 + b) * NOUT + o] = acc;
    }
}

extern "C" void kernel_launch(void* const* d_in, const int* in_sizes, int n_in,
                              void* d_out, int out_size, void* d_ws, size_t ws_size,
                              hipStream_t stream) {
    (void)in_sizes; (void)n_in; (void)d_ws; (void)ws_size; (void)out_size;
    const float* x    = (const float*)d_in[0];
    const float* Wih0 = (const float*)d_in[1];
    const float* Whh0 = (const float*)d_in[2];
    const float* bih0 = (const float*)d_in[3];
    const float* bhh0 = (const float*)d_in[4];
    const float* Wih1 = (const float*)d_in[5];
    const float* Whh1 = (const float*)d_in[6];
    const float* bih1 = (const float*)d_in[7];
    const float* bhh1 = (const float*)d_in[8];
    const float* Wfc  = (const float*)d_in[9];
    const float* bfc  = (const float*)d_in[10];
    float* out = (float*)d_out;

    dim3 grid(B_TOT / 16);   // 32 workgroups, one 16-row batch tile each
    dim3 block(256);         // 8 waves: one 16-col gate tile per wave
    lstm2_fused_wmma<<<grid, block, 0, stream>>>(x, Wih0, Whh0, bih0, bhh0,
                                                 Wih1, Whh1, bih1, bhh1,
                                                 Wfc, bfc, out);
}
